// GATGCNNet_60516089200775
// MI455X (gfx1250) — matrix operand order repaired and checked by reading, not compiled, not run on hardware
//
#include <hip/hip_runtime.h>

// ---------------------------------------------------------------------------
// GAT+GCN drug-pair network for MI455X (gfx1250, wave32, WMMA).
// Dense GEMMs: bf16 staging + v_wmma_f32_16x16x32_bf16, 2x2 register tiling
// (4 WMMAs / k-step per wave => 16 FLOP/byte of fragment traffic).
// Sparse phases: edge-parallel atomics (feature arrays fit in 192MB L2).
// ---------------------------------------------------------------------------

#define HH 10          // heads
#define FF 77          // per-head feature dim
#define CC 770         // HH*FF

// ---------------- device helpers ----------------

__device__ __forceinline__ unsigned short f2bf(float f) {
  unsigned int u = __float_as_uint(f);
  unsigned int r = u + 0x7FFFu + ((u >> 16) & 1u);   // round-to-nearest-even
  return (unsigned short)(r >> 16);
}
// order-preserving float -> uint key (for atomic max on floats)
__device__ __forceinline__ unsigned int fkey(float f) {
  unsigned int u = __float_as_uint(f);
  return (u & 0x80000000u) ? ~u : (u | 0x80000000u);
}
__device__ __forceinline__ float fkeyinv(unsigned int k) {
  unsigned int u = (k & 0x80000000u) ? (k & 0x7FFFFFFFu) : ~k;
  return __uint_as_float(u);
}

// ---------------- generic fill ----------------

__global__ void k_fill_u32(unsigned int* p, unsigned int v, size_t n) {
  size_t i = (size_t)blockIdx.x * blockDim.x + threadIdx.x;
  if (i < n) p[i] = v;
}

// ---------------- bf16 staging ----------------

// src [M,K] f32 row-major -> dst [M,Kp] bf16, zero pad K..Kp
__global__ void k_cvt_pad(const float* __restrict__ src, unsigned short* __restrict__ dst,
                          int M, int K, int Kp) {
  size_t i = (size_t)blockIdx.x * blockDim.x + threadIdx.x;
  if (i >= (size_t)M * Kp) return;
  int m = (int)(i / Kp), k = (int)(i % Kp);
  dst[i] = (k < K) ? f2bf(src[(size_t)m * K + k]) : (unsigned short)0;
}

// src [K,Nfull] f32 -> dst [Np,Kp] bf16 transposed, zero pad both dims
__global__ void k_cvt_pad_T(const float* __restrict__ src, unsigned short* __restrict__ dst,
                            int K, int Nfull, int Np, int Kp) {
  size_t i = (size_t)blockIdx.x * blockDim.x + threadIdx.x;
  if (i >= (size_t)Np * Kp) return;
  int n = (int)(i / Kp), k = (int)(i % Kp);
  dst[i] = (k < K && n < Nfull) ? f2bf(src[(size_t)k * Nfull + n]) : (unsigned short)0;
}

// ---------------- WMMA GEMM (2x2 wave tile) ----------------
// C[M,Nfull] = act( A[M,Kp](bf16) @ BT[Np,Kp]^T (bf16) + bias )
// Wave computes a 32x32 output tile = 4 accumulators; per k-step it loads
// 2 A-frags + 2 B-frags (8x b128) and issues 4 WMMAs. Block = 8 waves
// stacked in M (256-row block tile), so all waves share B frags via L0.
// Fragment layouts per CDNA5 ISA 7.12.2 (wave32):
//   A: lane L -> row m=L%16;   elems 0..7 = k0+8*(L/16)+j, 8..15 = k0+16+8*(L/16)+j
//   B: lane L -> col n=L%16;   elems 0..7 = k0+8*(L/16)+j, 8..15 = k0+16+8*(L/16)+j
//   C: vgpr r -> row r+8*(L/16), col L%16

__global__ __launch_bounds__(256) void k_gemm_wmma(
    const unsigned short* __restrict__ A, const unsigned short* __restrict__ BT,
    float* __restrict__ C, const float* __restrict__ bias,
    int M, int Nfull, int Kp, int act) {
#if defined(__HIP_DEVICE_COMPILE__)
  typedef __attribute__((ext_vector_type(16))) __bf16 v16bf;
  typedef __attribute__((ext_vector_type(8)))  float  v8f;

  int lane = threadIdx.x & 31;
  int wave = threadIdx.x >> 5;
  int m0 = blockIdx.y * 256 + wave * 32;
  if (m0 >= M) return;                       // wave-uniform guard (EXEC stays all-1 for WMMA)
  int n0 = blockIdx.x * 32;
  int half = lane >> 4;
  int lm = lane & 15;

  const unsigned short* arow0 = A  + (size_t)(m0 + lm) * Kp + half * 8;
  const unsigned short* arow1 = arow0 + (size_t)16 * Kp;
  const unsigned short* brow0 = BT + (size_t)(n0 + lm) * Kp + half * 8;
  const unsigned short* brow1 = brow0 + (size_t)16 * Kp;

  v8f c00 = {0.f,0.f,0.f,0.f,0.f,0.f,0.f,0.f};
  v8f c01 = c00, c10 = c00, c11 = c00;

  for (int k0 = 0; k0 < Kp; k0 += 32) {
    union { v16bf v; uint4 q[2]; } fa0, fa1, fb0, fb1;
    fa0.q[0] = *(const uint4*)(arow0 + k0);
    fa0.q[1] = *(const uint4*)(arow0 + k0 + 16);
    fa1.q[0] = *(const uint4*)(arow1 + k0);
    fa1.q[1] = *(const uint4*)(arow1 + k0 + 16);
    fb0.q[0] = *(const uint4*)(brow0 + k0);
    fb0.q[1] = *(const uint4*)(brow0 + k0 + 16);
    fb1.q[0] = *(const uint4*)(brow1 + k0);
    fb1.q[1] = *(const uint4*)(brow1 + k0 + 16);
    __builtin_prefetch(arow0 + k0 + 256, 0, 0);    // global_prefetch_b8, speculative
    __builtin_prefetch(brow0 + k0 + 256, 0, 0);
    c00 = __builtin_amdgcn_wmma_f32_16x16x32_bf16(false, fa0.v, false, fb0.v,
                                                  (short)0, c00, false, false);
    c01 = __builtin_amdgcn_wmma_f32_16x16x32_bf16(false, fa0.v, false, fb1.v,
                                                  (short)0, c01, false, false);
    c10 = __builtin_amdgcn_wmma_f32_16x16x32_bf16(false, fa1.v, false, fb0.v,
                                                  (short)0, c10, false, false);
    c11 = __builtin_amdgcn_wmma_f32_16x16x32_bf16(false, fa1.v, false, fb1.v,
                                                  (short)0, c11, false, false);
  }

  v8f accs[4] = {c00, c01, c10, c11};
#pragma unroll
  for (int t = 0; t < 4; t++) {
    int mt = t >> 1, nt = t & 1;
    int n = n0 + nt * 16 + lm;
    if (n < Nfull) {
      float bv = bias ? bias[n] : 0.0f;
#pragma unroll
      for (int r = 0; r < 8; r++) {
        int m = m0 + mt * 16 + r + half * 8;
        float v = accs[t][r] + bv;
        if (act == 1) v = fmaxf(v, 0.0f);
        C[(size_t)m * Nfull + n] = v;
      }
    }
  }
#endif
}

// ---------------- GAT kernels ----------------

// a_s[n,h] = sum_f xp[n,h,f]*att_s[h,f];  a_d likewise
__global__ void k_att(const float* __restrict__ xp, const float* __restrict__ as,
                      const float* __restrict__ ad, float* __restrict__ a_s,
                      float* __restrict__ a_d, int N) {
  int idx = blockIdx.x * blockDim.x + threadIdx.x;
  if (idx >= N * HH) return;
  int n = idx / HH, h = idx % HH;
  const float* xr = xp + (size_t)n * CC + h * FF;
  const float* s = as + h * FF;
  const float* d = ad + h * FF;
  float s1 = 0.f, s2 = 0.f;
  for (int f = 0; f < FF; f++) { float v = xr[f]; s1 += v * s[f]; s2 += v * d[f]; }
  a_s[idx] = s1; a_d[idx] = s2;
}

// pass 1: alpha = leaky_relu(a_s[src]+a_d[dst]); amax[dst] = max
__global__ void k_alpha_max(const int* __restrict__ ei, const float* __restrict__ a_s,
                            const float* __restrict__ a_d, float* __restrict__ alpha,
                            unsigned int* __restrict__ amax, int E, int E2) {
  int idx = blockIdx.x * blockDim.x + threadIdx.x;
  if (idx >= E2 * HH) return;
  int e = idx / HH, h = idx % HH;
  int s, d;
  if (e < E) { s = ei[e]; d = ei[E + e]; } else { s = d = e - E; }
  float al = a_s[s * HH + h] + a_d[d * HH + h];
  al = (al >= 0.f) ? al : 0.2f * al;
  alpha[idx] = al;
  atomicMax(&amax[d * HH + h], fkey(al));
}

// pass 2: e = exp(alpha - amax[dst]); denom[dst] += e
__global__ void k_exp_sum(const int* __restrict__ ei, float* __restrict__ alpha,
                          const unsigned int* __restrict__ amax, float* __restrict__ denom,
                          int E, int E2) {
  int idx = blockIdx.x * blockDim.x + threadIdx.x;
  if (idx >= E2 * HH) return;
  int e = idx / HH, h = idx % HH;
  int d = (e < E) ? ei[E + e] : (e - E);
  float m = fkeyinv(amax[d * HH + h]);
  float ev = expf(alpha[idx] - m);
  alpha[idx] = ev;
  atomicAdd(&denom[d * HH + h], ev);
}

// pass 3: out[dst, :] += xp[src, :] * w[h]   (one block per edge)
__global__ void k_gat_agg(const int* __restrict__ ei, const float* __restrict__ evals,
                          const float* __restrict__ denom, const float* __restrict__ xp,
                          float* __restrict__ out, int E) {
  int e = blockIdx.x;
  int s, d;
  if (e < E) { s = ei[e]; d = ei[E + e]; } else { s = d = e - E; }
  __shared__ float w[HH];
  if (threadIdx.x < HH)
    w[threadIdx.x] = evals[(size_t)e * HH + threadIdx.x] /
                     (denom[(size_t)d * HH + threadIdx.x] + 1e-16f);
  __syncthreads();
  const float* xr = xp + (size_t)s * CC;
  float* orow = out + (size_t)d * CC;
  for (int c = threadIdx.x; c < CC; c += blockDim.x)
    atomicAdd(&orow[c], xr[c] * w[c / FF]);
}

// ---------------- GCN kernels ----------------

__global__ void k_deg(const int* __restrict__ ei, float* __restrict__ deg, int E, int E2) {
  int e = blockIdx.x * blockDim.x + threadIdx.x;
  if (e >= E2) return;
  int d = (e < E) ? ei[E + e] : (e - E);
  atomicAdd(&deg[d], 1.0f);
}

__global__ void k_dinv(float* __restrict__ deg, int N) {
  int n = blockIdx.x * blockDim.x + threadIdx.x;
  if (n < N) deg[n] = rsqrtf(fmaxf(deg[n], 1.0f));
}

__global__ void k_gcn_agg(const int* __restrict__ ei, const float* __restrict__ h,
                          const float* __restrict__ dinv, float* __restrict__ out, int E) {
  int e = blockIdx.x;
  int s, d;
  if (e < E) { s = ei[e]; d = ei[E + e]; } else { s = d = e - E; }
  float w = dinv[s] * dinv[d];
  const float* hr = h + (size_t)s * CC;
  float* orow = out + (size_t)d * CC;
  for (int c = threadIdx.x; c < CC; c += blockDim.x)
    atomicAdd(&orow[c], hr[c] * w);
}

// ---------------- elementwise / pool / concat / head ----------------

// act: 0=none, 1=relu, 2=elu
__global__ void k_bias_act(float* __restrict__ x, const float* __restrict__ bias,
                           size_t total, int C, int act) {
  size_t i = (size_t)blockIdx.x * blockDim.x + threadIdx.x;
  if (i >= total) return;
  float v = x[i] + bias[i % C];
  if (act == 1) v = fmaxf(v, 0.0f);
  else if (act == 2) v = (v > 0.f) ? v : expm1f(v);
  x[i] = v;
}

// batch is contiguous: npg nodes per graph. pooled[b] = [max | mean], width 2*CC
__global__ void k_pool(const float* __restrict__ x, float* __restrict__ pooled,
                       int B, int npg) {
  int idx = blockIdx.x * blockDim.x + threadIdx.x;
  if (idx >= B * CC) return;
  int b = idx / CC, c = idx % CC;
  const float* p = x + (size_t)b * npg * CC + c;
  float mx = -INFINITY, sm = 0.f;
  for (int i = 0; i < npg; i++) { float v = p[(size_t)i * CC]; mx = fmaxf(mx, v); sm += v; }
  pooled[(size_t)b * (2 * CC) + c] = mx;
  pooled[(size_t)b * (2 * CC) + CC + c] = sm / (float)npg;
}

__global__ void k_concat4(float* __restrict__ dst,
                          const float* a, int wa, const float* b, int wb,
                          const float* c, int wc, const float* d, int wd, int rows) {
  int W = wa + wb + wc + wd;
  int idx = blockIdx.x * blockDim.x + threadIdx.x;
  if (idx >= rows * W) return;
  int r = idx / W, j = idx % W;
  float v;
  if (j < wa)                v = a[(size_t)r * wa + j];
  else if (j < wa + wb)      v = b[(size_t)r * wb + (j - wa)];
  else if (j < wa + wb + wc) v = c[(size_t)r * wc + (j - wa - wb)];
  else                       v = d[(size_t)r * wd + (j - wa - wb - wc)];
  dst[idx] = v;
}

// out[b] = clip( [f2|xtc] @ W + bias, -100, 100 )
__global__ void k_final(const float* __restrict__ f2, const float* __restrict__ xtc,
                        const float* __restrict__ W, const float* __restrict__ bias,
                        float* __restrict__ out, int B) {
  int r = blockIdx.x * blockDim.x + threadIdx.x;
  if (r >= B) return;
  float acc = bias[0];
  for (int j = 0; j < 256; j++) acc += f2[(size_t)r * 256 + j] * W[j];
  for (int j = 0; j < 10; j++)  acc += xtc[(size_t)r * 10 + j] * W[256 + j];
  out[r] = fminf(fmaxf(acc, -100.0f), 100.0f);
}

// ---------------- host orchestration ----------------

static inline unsigned cdiv(size_t a, unsigned b) { return (unsigned)((a + b - 1) / b); }
static inline int npad32(int n) { return (n + 31) & ~31; }

static void cvt_pad(const float* src, unsigned short* dst, int M, int K, int Kp, hipStream_t s) {
  size_t n = (size_t)M * Kp;
  k_cvt_pad<<<cdiv(n, 256), 256, 0, s>>>(src, dst, M, K, Kp);
}
static void cvt_pad_T(const float* src, unsigned short* dst, int K, int Nf, int Np, int Kp,
                      hipStream_t s) {
  size_t n = (size_t)Np * Kp;
  k_cvt_pad_T<<<cdiv(n, 256), 256, 0, s>>>(src, dst, K, Nf, Np, Kp);
}
static void rungemm(const unsigned short* A, const unsigned short* BT, float* C,
                    const float* bias, int M, int Nf, int Np, int Kp, int act, hipStream_t s) {
  dim3 grid((unsigned)(Np / 32), (unsigned)((M + 255) / 256));
  k_gemm_wmma<<<grid, 256, 0, s>>>(A, BT, C, bias, M, Nf, Kp, act);
}

extern "C" void kernel_launch(void* const* d_in, const int* in_sizes, int n_in,
                              void* d_out, int out_size, void* d_ws, size_t ws_size,
                              hipStream_t stream) {
  (void)n_in; (void)out_size; (void)ws_size;
  const int N   = in_sizes[0] / FF;    // 20480 nodes
  const int E   = in_sizes[1] / 2;     // 81920 edges
  const int B   = in_sizes[27] / 10;   // 512 graphs
  const int NPG = N / B;               // 40 nodes/graph
  const int E2  = E + N;               // edges incl. self loops

  char* ws = (char*)d_ws;
  size_t off = 0;
  auto alloc = [&](size_t bytes) -> char* {
    char* p = ws + off;
    off += (bytes + 255) & ~(size_t)255;
    return p;
  };

  float*          xp     = (float*)alloc((size_t)N * CC * 4);      // GAT lin out / GCN lin out
  float*          big2   = (float*)alloc((size_t)N * CC * 4);      // aggregation target
  unsigned short* stageA = (unsigned short*)alloc((size_t)N * 800 * 2);
  unsigned short* stageW = (unsigned short*)alloc((size_t)1504 * 1568 * 2);
  float*          a_s    = (float*)alloc((size_t)N * HH * 4);
  float*          a_d    = (float*)alloc((size_t)N * HH * 4);
  float*          alpha  = (float*)alloc((size_t)E2 * HH * 4);
  unsigned int*   amax   = (unsigned int*)alloc((size_t)N * HH * 4);
  float*          denom  = (float*)alloc((size_t)N * HH * 4);
  float*          deg    = (float*)alloc((size_t)N * 4);
  float*          pooled = (float*)alloc((size_t)B * 1540 * 4);
  float*          mid1   = (float*)alloc((size_t)B * 2048 * 4);
  float*          mid2   = (float*)alloc((size_t)B * 256 * 4);
  float*          h1     = (float*)alloc((size_t)B * 128 * 4);
  float*          h2     = (float*)alloc((size_t)B * 128 * 4);
  float*          catb   = (float*)alloc((size_t)B * 1546 * 4);

  auto fill = [&](void* p, unsigned int v, size_t n) {
    k_fill_u32<<<cdiv(n, 256), 256, 0, stream>>>((unsigned int*)p, v, n);
  };
  const unsigned int NEG_INF_KEY = 0x007FFFFFu;  // fkey(-inf)

  const float* xtc = (const float*)d_in[27];

  // ---------------- drug branches ----------------
  for (int dbr = 0; dbr < 2; dbr++) {
    int ib = dbr * 11;
    const float* xd    = (const float*)d_in[ib + 0];
    const int*   ei    = (const int*)  d_in[ib + 1];
    const float* gatW  = (const float*)d_in[ib + 3];
    const float* atts  = (const float*)d_in[ib + 4];
    const float* attd  = (const float*)d_in[ib + 5];
    const float* gatb  = (const float*)d_in[ib + 6];
    const float* gcnW  = (const float*)d_in[ib + 7];
    const float* gcnb  = (const float*)d_in[ib + 8];
    const float* fg1W  = (const float*)d_in[ib + 9];
    const float* fg1b  = (const float*)d_in[ib + 10];
    const float* fg2W  = (const float*)d_in[22];
    const float* fg2b  = (const float*)d_in[23];
    float* hd = (dbr == 0) ? h1 : h2;

    // GAT linear: xp = x @ gat_W   [N,77]x[77,770]
    cvt_pad(xd, stageA, N, 77, 96, stream);
    cvt_pad_T(gatW, stageW, 77, 770, npad32(770), 96, stream);
    rungemm(stageA, stageW, xp, nullptr, N, 770, npad32(770), 96, 0, stream);

    // attention scores
    k_att<<<cdiv((size_t)N * HH, 256), 256, 0, stream>>>(xp, atts, attd, a_s, a_d, N);

    // softmax over incoming edges (self loops guarantee finite max)
    fill(amax, NEG_INF_KEY, (size_t)N * HH);
    fill(denom, 0u, (size_t)N * HH);
    fill(big2, 0u, (size_t)N * CC);
    fill(deg, 0u, (size_t)N);
    k_alpha_max<<<cdiv((size_t)E2 * HH, 256), 256, 0, stream>>>(ei, a_s, a_d, alpha, amax, E, E2);
    k_exp_sum<<<cdiv((size_t)E2 * HH, 256), 256, 0, stream>>>(ei, alpha, amax, denom, E, E2);
    k_gat_agg<<<(unsigned)E2, 256, 0, stream>>>(ei, alpha, denom, xp, big2, E);
    k_bias_act<<<cdiv((size_t)N * CC, 256), 256, 0, stream>>>(big2, gatb, (size_t)N * CC, CC, 2); // +b, ELU

    // GCN: degree norm
    k_deg<<<cdiv((size_t)E2, 256), 256, 0, stream>>>(ei, deg, E, E2);
    k_dinv<<<cdiv((size_t)N, 256), 256, 0, stream>>>(deg, N);

    // GCN linear: h = x1 @ gcn_W   [N,770]x[770,770]  (dominant GEMM)
    cvt_pad(big2, stageA, N, 770, 800, stream);
    cvt_pad_T(gcnW, stageW, 770, 770, npad32(770), 800, stream);
    rungemm(stageA, stageW, xp, nullptr, N, 770, npad32(770), 800, 0, stream);

    // normalized aggregation + bias + relu
    fill(big2, 0u, (size_t)N * CC);
    k_gcn_agg<<<(unsigned)E2, 256, 0, stream>>>(ei, xp, deg, big2, E);
    k_bias_act<<<cdiv((size_t)N * CC, 256), 256, 0, stream>>>(big2, gcnb, (size_t)N * CC, CC, 1);

    // pool max|mean -> [B,1540]
    k_pool<<<cdiv((size_t)B * CC, 256), 256, 0, stream>>>(big2, pooled, B, NPG);

    // fcg1: [B,1540]x[1540,1500] + b, relu
    cvt_pad(pooled, stageA, B, 1540, 1568, stream);
    cvt_pad_T(fg1W, stageW, 1540, 1500, npad32(1500), 1568, stream);
    rungemm(stageA, stageW, mid1, fg1b, B, 1500, npad32(1500), 1568, 1, stream);

    // fcg2 (shared): [B,1500]x[1500,128] + b, relu -> h_d
    cvt_pad(mid1, stageA, B, 1500, 1504, stream);
    cvt_pad_T(fg2W, stageW, 1500, 128, 128, 1504, stream);
    rungemm(stageA, stageW, hd, fg2b, B, 128, 128, 1504, 1, stream);
  }

  // ---------------- cell-line branch ----------------
  const float* xc1  = (const float*)d_in[24];
  const float* xc2  = (const float*)d_in[25];
  const float* xc3  = (const float*)d_in[26];
  const float* clW1 = (const float*)d_in[28];
  const float* clb1 = (const float*)d_in[29];
  const float* clW2 = (const float*)d_in[30];
  const float* clb2 = (const float*)d_in[31];

  k_concat4<<<cdiv((size_t)B * 1546, 256), 256, 0, stream>>>(catb, xc1, 512, xc2, 512, xc3, 512, xtc, 10, B);
  cvt_pad(catb, stageA, B, 1546, 1568, stream);
  cvt_pad_T(clW1, stageW, 1546, 1024, 1024, 1568, stream);
  rungemm(stageA, stageW, mid1, clb1, B, 1024, 1024, 1568, 1, stream);
  cvt_pad(mid1, stageA, B, 1024, 1024, stream);
  cvt_pad_T(clW2, stageW, 1024, 256, 256, 1024, stream);
  rungemm(stageA, stageW, mid2, clb2, B, 256, 256, 1024, 1, stream);   // xcl

  // ---------------- head ----------------
  const float* fc1W = (const float*)d_in[32];
  const float* fc1b = (const float*)d_in[33];
  const float* fc2W = (const float*)d_in[34];
  const float* fc2b = (const float*)d_in[35];
  const float* outW = (const float*)d_in[36];
  const float* outb = (const float*)d_in[37];

  k_concat4<<<cdiv((size_t)B * 522, 256), 256, 0, stream>>>(catb, h1, 128, h2, 128, mid2, 256, xtc, 10, B);
  cvt_pad(catb, stageA, B, 522, 544, stream);
  cvt_pad_T(fc1W, stageW, 522, 2048, 2048, 544, stream);
  rungemm(stageA, stageW, mid1, fc1b, B, 2048, 2048, 544, 1, stream);
  cvt_pad(mid1, stageA, B, 2048, 2048, stream);
  cvt_pad_T(fc2W, stageW, 2048, 256, 256, 2048, stream);
  rungemm(stageA, stageW, mid2, fc2b, B, 256, 256, 2048, 1, stream);   // f2 (reuse mid2)

  k_final<<<cdiv((size_t)B, 256), 256, 0, stream>>>(mid2, xtc, outW, outb, (float*)d_out, B);
}
